// MultiHeadAttention_33835752358395
// MI455X (gfx1250) — matrix operand order repaired and checked by reading, not compiled, not run on hardware
//
#include <hip/hip_runtime.h>
#include <math.h>

#define D_MODEL 1024
#define N_HEAD  16
#define HEAD    64
#define BATCH   2
#define SEQ     2048
#define MROWS   (BATCH * SEQ)   // 4096

typedef _Float16 h8  __attribute__((ext_vector_type(8)));
typedef _Float16 h16 __attribute__((ext_vector_type(16)));
typedef float    f8  __attribute__((ext_vector_type(8)));

#if __has_builtin(__builtin_amdgcn_tensor_load_to_lds)
#define HAVE_TDM 1
#else
#define HAVE_TDM 0
#endif

__device__ __forceinline__ h16 cat8(h8 lo, h8 hi) {
  h16 r;
#pragma unroll
  for (int i = 0; i < 8; ++i) { r[i] = lo[i]; r[8 + i] = hi[i]; }
  return r;
}

// A fragment (16x32 f16, M x K). Layout: lanes 0-15 hold row M=lane,
// K = 0..15; lanes 16-31 hold K = 16..31.
__device__ __forceinline__ h16 load_a(const _Float16* __restrict__ p, int ld,
                                      int row0, int k0, int lane) {
  int hf = lane >> 4, lm = lane & 15;
  const _Float16* r0 = p + (row0 + lm) * ld + k0 + hf * 16;
  h8 lo = *(const h8*)(r0);
  h8 hi = *(const h8*)(r0 + 8);
  return cat8(lo, hi);
}

// B fragment (32x16 f16, K x N) where B[k][n] = p[n*ld + k] (contiguous in k).
// Contiguous k order (same as load_a): lanes 0-15 hold N=lane, K=0..15; lanes 16-31 hold K=16..31.
__device__ __forceinline__ h16 load_b(const _Float16* __restrict__ p, int ld,
                                      int n0, int k0, int lane) {
  int hf = lane >> 4, lm = lane & 15;
  const _Float16* r0 = p + (n0 + lm) * ld + k0 + hf * 16;
  h8 lo = *(const h8*)(r0);
  h8 hi = *(const h8*)(r0 + 8);
  return cat8(lo, hi);
}

// B fragment from a row-major [64][32] f16 LDS tile.
__device__ __forceinline__ h16 load_b_lds(const _Float16* buf, int j, int hf, int lm) {
  const _Float16* r0 = buf + (j * 16 + lm) * 32 + hf * 16;
  h8 lo = *(const h8*)(r0);
  h8 hi = *(const h8*)(r0 + 8);
  return cat8(lo, hi);
}

#if HAVE_TDM
typedef unsigned int u32x4 __attribute__((ext_vector_type(4)));
typedef int          i32x8 __attribute__((ext_vector_type(8)));
typedef int          i32x4 __attribute__((ext_vector_type(4)));

// TDM descriptor: load a 32x64 tile (tile_dim0=32 contiguous f16, 64 rows,
// row stride 1024 f16) from W into LDS, row-major [64][32].
__device__ __forceinline__ void tdm_load_w_tile(const _Float16* gptr, void* lds_dst) {
  unsigned long long ga = (unsigned long long)(uintptr_t)gptr;
  unsigned int laddr = (unsigned int)(uintptr_t)lds_dst;  // low 32b = LDS offset
  u32x4 g0;
  g0[0] = 1u;                                          // count=1, no gather
  g0[1] = laddr;                                       // lds_addr
  g0[2] = (unsigned int)ga;                            // global_addr[31:0]
  g0[3] = (unsigned int)((ga >> 32) & 0x1FFFFFFu) | (2u << 30);  // ga[56:32], type=2
  i32x8 g1;
  g1[0] = (int)(1u << 16);        // data_size = 1 (2 bytes)
  g1[1] = (int)(1024u << 16);     // tensor_dim0[15:0] = 1024
  g1[2] = (int)(1024u << 16);     // tensor_dim0 hi=0 | tensor_dim1[15:0] = 1024
  g1[3] = (int)(32u << 16);       // tensor_dim1 hi=0 | tile_dim0 = 32
  g1[4] = 64;                     // tile_dim1 = 64, tile_dim2 = 0
  g1[5] = 1024;                   // tensor_dim0_stride[31:0] = 1024
  g1[6] = 0;
  g1[7] = 0;
  i32x4 z4 = {0, 0, 0, 0};
  i32x8 z8 = {0, 0, 0, 0, 0, 0, 0, 0};
  // 6-arg form (clang-23 / therock-10.0 headers)
  __builtin_amdgcn_tensor_load_to_lds(g0, g1, z4, z4, z8, 0);
}
#endif

// ---------------------------------------------------------------- convert ---
__global__ __launch_bounds__(256) void cvt_f16(const float* __restrict__ src,
                                               _Float16* __restrict__ dst, int n8) {
  int i = blockIdx.x * blockDim.x + threadIdx.x;
  if (i >= n8) return;
  const float4* s = (const float4*)src;
  float4 a = s[2 * i], b = s[2 * i + 1];
  h8 o;
  o[0] = (_Float16)a.x; o[1] = (_Float16)a.y; o[2] = (_Float16)a.z; o[3] = (_Float16)a.w;
  o[4] = (_Float16)b.x; o[5] = (_Float16)b.y; o[6] = (_Float16)b.z; o[7] = (_Float16)b.w;
  *(h8*)(dst + 8 * i) = o;
}

// --------------------------------------------------- block-tiled GEMM -------
// Y = X @ W^T. Block = 256 rows x 64 cols (8 waves x 32 rows). W tile staged
// in LDS, double-buffered via TDM (wave 0 issues, s_wait_tensorcnt + barrier).
// mode 0: f16 out [B,H,T,Dh];  mode 1: f16 out [B,H,Dh,T];  mode 2: f32 row-major.
__global__ __launch_bounds__(256) void gemm_block(const _Float16* __restrict__ X,
                                                  const _Float16* __restrict__ W,
                                                  _Float16* __restrict__ out16,
                                                  float* __restrict__ outf,
                                                  int mode, float scale) {
  __shared__ _Float16 ldsW[2][64 * 32];   // 2 x 4 KB double buffer
  int wlocal = threadIdx.x >> 5;
  int lane = threadIdx.x & 31;
  int mb = blockIdx.x >> 4, nb = blockIdx.x & 15;   // 16 x 16 blocks
  int m0 = mb * 256 + wlocal * 32, n0 = nb * 64;
  int hf = lane >> 4, lm = lane & 15;
  f8 acc[2][4] = {};

#if HAVE_TDM
  if (wlocal == 0) {
    tdm_load_w_tile(W + n0 * D_MODEL + 0,  &ldsW[0][0]);
    tdm_load_w_tile(W + n0 * D_MODEL + 32, &ldsW[1][0]);
  }
#endif

  for (int kb = 0; kb < 32; ++kb) {
    int k0 = kb * 32;
    int cur = kb & 1;
#if HAVE_TDM
    if (wlocal == 0) {
      if (kb < 31) __builtin_amdgcn_s_wait_tensorcnt(1);
      else         __builtin_amdgcn_s_wait_tensorcnt(0);
    }
    __syncthreads();                       // tile kb visible to all waves
    const _Float16* bufp = &ldsW[cur][0];
#else
    __syncthreads();                       // previous readers done
    {
      int t = threadIdx.x;
      int row = t >> 2;                    // 64 rows, 4 threads/row
      int col = (t & 3) * 8;               // 8 halves each
      *(h8*)&ldsW[0][row * 32 + col] =
          *(const h8*)&W[(n0 + row) * D_MODEL + k0 + col];
    }
    __syncthreads();
    const _Float16* bufp = &ldsW[0][0];
#endif

    h16 a0 = load_a(X, D_MODEL, m0,      k0, lane);
    h16 a1 = load_a(X, D_MODEL, m0 + 16, k0, lane);
#pragma unroll
    for (int j = 0; j < 4; ++j) {
      h16 b = load_b_lds(bufp, j, hf, lm);
      acc[0][j] = __builtin_amdgcn_wmma_f32_16x16x32_f16(false, a0, false, b,
                                                         (short)0, acc[0][j], false, false);
      acc[1][j] = __builtin_amdgcn_wmma_f32_16x16x32_f16(false, a1, false, b,
                                                         (short)0, acc[1][j], false, false);
    }

    __syncthreads();                       // all reads of buf[cur] done
#if HAVE_TDM
    if (wlocal == 0 && kb + 2 < 32)
      tdm_load_w_tile(W + n0 * D_MODEL + (k0 + 64), &ldsW[cur][0]);
#endif
  }

#pragma unroll
  for (int i = 0; i < 2; ++i)
#pragma unroll
    for (int j = 0; j < 4; ++j) {
      int n = n0 + j * 16 + lm;
#pragma unroll
      for (int r = 0; r < 8; ++r) {
        int m = m0 + i * 16 + r + 8 * hf;
        if (mode == 2) {
          outf[m * D_MODEL + n] = acc[i][j][r];
        } else {
          int h = n >> 6, d = n & 63;
          int bbb = m >> 11, t = m & (SEQ - 1);
          int idx = (mode == 1) ? ((bbb * N_HEAD + h) * HEAD + d) * SEQ + t
                                : ((bbb * N_HEAD + h) * SEQ + t) * HEAD + d;
          out16[idx] = (_Float16)(acc[i][j][r] * scale);
        }
      }
    }
}

// ------------------------------------------------- flash attention (causal) --
__global__ __launch_bounds__(256) void attn(const _Float16* __restrict__ Qh,
                                            const _Float16* __restrict__ Kh,
                                            const _Float16* __restrict__ VhT,
                                            _Float16* __restrict__ Aout) {
  __shared__ _Float16 ldsP[8][16 * 32];   // per-wave P staging tile (1 KB each)
  int wave   = (int)((blockIdx.x * blockDim.x + threadIdx.x) >> 5);
  int wlocal = threadIdx.x >> 5;
  int lane   = threadIdx.x & 31;
  int bh = wave >> 7;          // 32 (b,h) pairs
  int qt = wave & 127;         // 128 query tiles
  int bb = bh >> 4, h = bh & 15;
  int m0 = qt * 16;
  int hf = lane >> 4, lm = lane & 15;

  const _Float16* Qbase = Qh  + ((bb * N_HEAD + h) * SEQ) * HEAD;
  const _Float16* Kbase = Kh  + ((bb * N_HEAD + h) * SEQ) * HEAD;
  const _Float16* Vbase = VhT + ((bb * N_HEAD + h) * HEAD) * SEQ;

  h16 aQ[2];
#pragma unroll
  for (int dc = 0; dc < 2; ++dc) aQ[dc] = load_a(Qbase, HEAD, m0, dc * 32, lane);

  f8 acc[4] = {};
  float mrow[8], lrow[8];
#pragma unroll
  for (int r = 0; r < 8; ++r) { mrow[r] = -__builtin_inff(); lrow[r] = 0.f; }

  int nblk = (m0 + 16 + 31) >> 5;          // causal: only key blocks <= diag
  for (int kb = 0; kb < nblk; ++kb) {
    int k0 = kb * 32;

    // issue all K fragments for this block, then the 4 S-WMMAs (single wait)
    h16 bk[2][2];
#pragma unroll
    for (int dc = 0; dc < 2; ++dc)
#pragma unroll
      for (int jj = 0; jj < 2; ++jj)
        bk[dc][jj] = load_b(Kbase, HEAD, k0 + jj * 16, dc * 32, lane);

    f8 s[2] = {};
#pragma unroll
    for (int dc = 0; dc < 2; ++dc)
#pragma unroll
      for (int jj = 0; jj < 2; ++jj)
        s[jj] = __builtin_amdgcn_wmma_f32_16x16x32_f16(false, aQ[dc], false, bk[dc][jj],
                                                       (short)0, s[jj], false, false);

    // issue V fragments NOW so their latency hides under the softmax VALU work
    h16 bv[4];
#pragma unroll
    for (int j = 0; j < 4; ++j)
      bv[j] = load_b(Vbase, SEQ, j * 16, k0, lane);

    // causal mask + online softmax (rows live across the 16-lane N group)
    float pr[2][8];
#pragma unroll
    for (int r = 0; r < 8; ++r) {
      int q = m0 + r + 8 * hf;
      float s0 = (k0 + lm      > q) ? -__builtin_inff() : s[0][r];
      float s1 = (k0 + 16 + lm > q) ? -__builtin_inff() : s[1][r];
      float tm = fmaxf(s0, s1);
      tm = fmaxf(tm, __shfl_xor(tm, 1, 32));
      tm = fmaxf(tm, __shfl_xor(tm, 2, 32));
      tm = fmaxf(tm, __shfl_xor(tm, 4, 32));
      tm = fmaxf(tm, __shfl_xor(tm, 8, 32));
      float newm = fmaxf(mrow[r], tm);
      float sf = __expf(mrow[r] - newm);
      float p0 = __expf(s0 - newm);
      float p1 = __expf(s1 - newm);
      float ps = p0 + p1;
      ps += __shfl_xor(ps, 1, 32);
      ps += __shfl_xor(ps, 2, 32);
      ps += __shfl_xor(ps, 4, 32);
      ps += __shfl_xor(ps, 8, 32);
      lrow[r] = lrow[r] * sf + ps;
      mrow[r] = newm;
#pragma unroll
      for (int j = 0; j < 4; ++j) acc[j][r] *= sf;
      pr[0][r] = p0; pr[1][r] = p1;
    }

    // stage P (C layout) through LDS, reload in A-fragment layout
    _Float16* P = &ldsP[wlocal][0];
#pragma unroll
    for (int jj = 0; jj < 2; ++jj)
#pragma unroll
      for (int r = 0; r < 8; ++r)
        P[(r + 8 * hf) * 32 + jj * 16 + lm] = (_Float16)pr[jj][r];
    asm volatile("s_wait_dscnt 0" ::: "memory");   // wave-local DS fence

    const _Float16* prow = P + lm * 32 + hf * 16;
    h8 plo = *(const h8*)(prow);
    h8 phi = *(const h8*)(prow + 8);
    h16 aP = cat8(plo, phi);

    // O += P @ V   (V fragments already in flight/registers)
#pragma unroll
    for (int j = 0; j < 4; ++j)
      acc[j] = __builtin_amdgcn_wmma_f32_16x16x32_f16(false, aP, false, bv[j],
                                                      (short)0, acc[j], false, false);
    asm volatile("s_wait_dscnt 0" ::: "memory");   // reads done before next stores
  }

  // normalize + store attention output as f16 in [B,T,E]
#pragma unroll
  for (int r = 0; r < 8; ++r) {
    float inv = 1.f / lrow[r];
    int t = m0 + r + 8 * hf;
#pragma unroll
    for (int j = 0; j < 4; ++j) {
      int e = h * HEAD + j * 16 + lm;
      Aout[(bb * SEQ + t) * D_MODEL + e] = (_Float16)(acc[j][r] * inv);
    }
  }
}

// ---------------------------------------------------------------- launch ----
extern "C" void kernel_launch(void* const* d_in, const int* in_sizes, int n_in,
                              void* d_out, int out_size, void* d_ws, size_t ws_size,
                              hipStream_t stream) {
  const float* q  = (const float*)d_in[0];
  const float* k  = (const float*)d_in[1];
  const float* v  = (const float*)d_in[2];
  const float* Wq = (const float*)d_in[3];
  const float* Wk = (const float*)d_in[4];
  const float* Wv = (const float*)d_in[5];
  const float* Wo = (const float*)d_in[6];

  const int NX = MROWS * D_MODEL;      // 4M elements per activation
  const int NW = D_MODEL * D_MODEL;    // 1M elements per weight

  _Float16* ws   = (_Float16*)d_ws;    // 32M f16 elements = 64 MB total
  _Float16* q16  = ws;
  _Float16* k16  = q16 + NX;
  _Float16* v16  = k16 + NX;
  _Float16* wq16 = v16 + NX;
  _Float16* wk16 = wq16 + NW;
  _Float16* wv16 = wk16 + NW;
  _Float16* wo16 = wv16 + NW;
  _Float16* Qh   = wo16 + NW;          // [B,H,T,Dh]
  _Float16* Kh   = Qh + NX;            // [B,H,T,Dh]
  _Float16* VhT  = Kh + NX;            // [B,H,Dh,T]
  _Float16* aout = VhT + NX;           // [B,T,E]

  // fp32 -> f16 conversions
  cvt_f16<<<(NX / 8 + 255) / 256, 256, 0, stream>>>(q, q16, NX / 8);
  cvt_f16<<<(NX / 8 + 255) / 256, 256, 0, stream>>>(k, k16, NX / 8);
  cvt_f16<<<(NX / 8 + 255) / 256, 256, 0, stream>>>(v, v16, NX / 8);
  cvt_f16<<<(NW / 8 + 255) / 256, 256, 0, stream>>>(Wq, wq16, NW / 8);
  cvt_f16<<<(NW / 8 + 255) / 256, 256, 0, stream>>>(Wk, wk16, NW / 8);
  cvt_f16<<<(NW / 8 + 255) / 256, 256, 0, stream>>>(Wv, wv16, NW / 8);
  cvt_f16<<<(NW / 8 + 255) / 256, 256, 0, stream>>>(Wo, wo16, NW / 8);

  // projections: 16 row-blocks x 16 col-blocks = 256 blocks of 256 threads
  gemm_block<<<256, 256, 0, stream>>>(q16, wq16, Qh,  nullptr, 0, 0.125f); // 1/sqrt(64)
  gemm_block<<<256, 256, 0, stream>>>(k16, wk16, Kh,  nullptr, 0, 1.0f);
  gemm_block<<<256, 256, 0, stream>>>(v16, wv16, VhT, nullptr, 1, 1.0f);

  // causal flash attention: 32 (b,h) * 128 query tiles = 4096 waves
  attn<<<512, 256, 0, stream>>>(Qh, Kh, VhT, aout);

  // output projection to fp32
  gemm_block<<<256, 256, 0, stream>>>(aout, wo16, nullptr, (float*)d_out, 2, 1.0f);
}
